// Text_Texture_Erase_And_Enhance_Module_53480932770080
// MI455X (gfx1250) — matrix operand order repaired and checked by reading, not compiled, not run on hardware
//
#include <hip/hip_runtime.h>
#include <hip/hip_bf16.h>

// Text_Texture_Erase_And_Enhance module for MI455X (gfx1250, wave32).
// Attention GEMMs use v_wmma_f32_16x16x32_f16 (f16 A/B, f32 accumulate).
// Workspace requirement: ~275 MB.

typedef __attribute__((ext_vector_type(16))) _Float16 v16h;
typedef __attribute__((ext_vector_type(8)))  _Float16 v8h;
typedef __attribute__((ext_vector_type(8)))  float    v8f;

#define DEV __device__ __forceinline__

// ---------------------------------------------------------------- WMMA helpers
// A fragment (16x32, M x K), matrix row-major with leading dim `ld` (halves).
// A layout: lane l -> row l&15; halves h -> K = (l>>4)*16 + h (matches B).
DEV v16h load_A_frag(const _Float16* __restrict__ base, int ld, int r0, int k0) {
  int l  = threadIdx.x & 31;
  int r  = l & 15;
  int hi = l >> 4;
  const _Float16* p = base + (size_t)(r0 + r) * ld + k0 + hi * 16;
  return *(const v16h*)p;           // K = k0 + hi*16 + [0..15]
}

// B fragment (32x16, K x N). B[k][j] is read from a K-contiguous array:
// element (k, j) = base[(r0 + j)*ld + k0 + k]. Lane l -> col l&15,
// halves h -> K = (l>>4)*16 + h  => one aligned 32B chunk per lane.
DEV v16h load_B_frag(const _Float16* __restrict__ base, int ld, int r0, int k0) {
  int l  = threadIdx.x & 31;
  int j  = l & 15;
  int hi = l >> 4;
  const _Float16* p = base + (size_t)(r0 + j) * ld + k0 + hi * 16;
  return *(const v16h*)p;
}

DEV v8f wmma_f16(v16h a, v16h b, v8f c) {
  return __builtin_amdgcn_wmma_f32_16x16x32_f16(
      /*neg_a=*/false, a, /*neg_b=*/false, b,
      /*c_mod=*/(short)0, c, /*reuse_a=*/false, /*reuse_b=*/false);
}

DEV float sigmoidf(float x) { return 1.f / (1.f + __expf(-x)); }

// ---------------------------------------------------------------- pool kernels
__global__ void k_maxpool4(const float* __restrict__ in, float* __restrict__ out,
                           int CT, int h, int w) {           // in [CT,4h,4w]
  int idx = blockIdx.x * 256 + threadIdx.x;
  if (idx >= CT * h * w) return;
  int x = idx % w, y = (idx / w) % h, c = idx / (w * h);
  const float* p = in + ((size_t)c * (4 * h) + 4 * y) * (4 * w) + 4 * x;
  float v = -3.4e38f;
#pragma unroll
  for (int dy = 0; dy < 4; ++dy)
#pragma unroll
    for (int dx = 0; dx < 4; ++dx) v = fmaxf(v, p[(size_t)dy * 4 * w + dx]);
  out[idx] = v;
}

__global__ void k_maxpool2(const float* __restrict__ in, float* __restrict__ out,
                           int CT, int h, int w) {           // in [CT,2h,2w]
  int idx = blockIdx.x * 256 + threadIdx.x;
  if (idx >= CT * h * w) return;
  int x = idx % w, y = (idx / w) % h, c = idx / (w * h);
  const float* p = in + ((size_t)c * (2 * h) + 2 * y) * (2 * w) + 2 * x;
  out[idx] = fmaxf(fmaxf(p[0], p[1]), fmaxf(p[2 * w], p[2 * w + 1]));
}

// ---------------------------------------------------------------- 1x1 conv
__global__ void k_conv1x1(const float* __restrict__ in, const float* __restrict__ w,
                          const float* __restrict__ bias, float* __restrict__ out,
                          int Cin, int Cout, int HW) {
  int p = blockIdx.x * 256 + threadIdx.x;
  if (p >= HW) return;
  int bc = blockIdx.y, b = bc / Cout, co = bc % Cout;
  const float* ip = in + (size_t)b * Cin * HW + p;
  const float* wp = w + (size_t)co * Cin;    // wave-uniform -> s_loads
  float acc = bias[co];
  for (int ci = 0; ci < Cin; ++ci) acc += wp[ci] * ip[(size_t)ci * HW];
  out[(size_t)bc * HW + p] = acc;
}

// f1 = conv1x1(cat[x1_1,x2_1,x3], wc) ; g0 = cat[f1*mask, mask]  -> [B,65,4096]
__global__ void k_f1(const float* __restrict__ a, const float* __restrict__ c2,
                     const float* __restrict__ c3, const float* __restrict__ wc,
                     const float* __restrict__ bc, const float* __restrict__ mask,
                     float* __restrict__ g) {
  int p = blockIdx.x * 256 + threadIdx.x;          // 4096 pixels
  int bc_ = blockIdx.y, b = bc_ / 65, co = bc_ % 65;
  float mv = mask[b * 4096 + p];
  size_t ob = (size_t)bc_ * 4096 + p;
  if (co == 64) { g[ob] = mv; return; }
  const float* w0 = wc + (size_t)co * 192;
  const float* pa = a  + (size_t)b * 64 * 4096 + p;
  const float* pb = c2 + (size_t)b * 64 * 4096 + p;
  const float* pc = c3 + (size_t)b * 64 * 4096 + p;
  float acc = bc[co];
  for (int ci = 0; ci < 64; ++ci) acc += w0[ci]       * pa[(size_t)ci * 4096];
  for (int ci = 0; ci < 64; ++ci) acc += w0[64 + ci]  * pb[(size_t)ci * 4096];
  for (int ci = 0; ci < 64; ++ci) acc += w0[128 + ci] * pc[(size_t)ci * 4096];
  g[ob] = acc * mv;
}

// ---------------------------------------------------------------- gate convs
__global__ void k_conv_direct(const float* __restrict__ in, const float* __restrict__ w,
                              const float* __restrict__ bias, float* __restrict__ out,
                              int K, int pad) {     // 65 -> 65 ch on 64x64
  int p = blockIdx.x * 256 + threadIdx.x;
  int py = p >> 6, px = p & 63;
  int bc_ = blockIdx.y, b = bc_ / 65, co = bc_ % 65;
  const float* wb = w  + (size_t)co * 65 * K * K;   // wave-uniform
  const float* ib = in + (size_t)b * 65 * 4096;
  float acc = bias[co];
  for (int ci = 0; ci < 65; ++ci) {
    const float* ic = ib + (size_t)ci * 4096;
    const float* wk = wb + (size_t)ci * K * K;
    for (int ky = 0; ky < K; ++ky) {
      int iy = py + ky - pad;
      if ((unsigned)iy >= 64u) continue;
      for (int kx = 0; kx < K; ++kx) {
        int ix = px + kx - pad;
        if ((unsigned)ix >= 64u) continue;
        acc = fmaf(wk[ky * K + kx], ic[iy * 64 + ix], acc);
      }
    }
  }
  out[(size_t)bc_ * 4096 + p] = acc;
}

__global__ void k_gate(const float* __restrict__ y, float* __restrict__ g) {
  int p = blockIdx.x * 256 + threadIdx.x;
  int bc_ = blockIdx.y, b = bc_ / 65, co = bc_ % 65;
  float m = sigmoidf(y[((size_t)b * 65 + 64) * 4096 + p]);
  float o = (co == 64) ? m : fmaxf(y[(size_t)bc_ * 4096 + p], 0.f) * m;
  g[(size_t)bc_ * 4096 + p] = o;
}

// ---------------------------------------------------------------- f16 staging
__global__ void k_to_half(const float* __restrict__ in, _Float16* __restrict__ out, int n) {
  int i = blockIdx.x * 256 + threadIdx.x;
  if (i < n) out[i] = (_Float16)in[i];
}
// fT[b][n][c] = g[b][c][n] (c < 64 of the 65-ch gate output)
__global__ void k_fT(const float* __restrict__ g, _Float16* __restrict__ fT) {
  int i = blockIdx.x * 256 + threadIdx.x;       // 2*4096*64
  int c = i & 63, n = (i >> 6) & 4095, b = i >> 18;
  fT[i] = (_Float16)g[((size_t)b * 65 + c) * 4096 + n];
}
// fh[b][c][n] = g[b][c][n]
__global__ void k_fh(const float* __restrict__ g, _Float16* __restrict__ fh) {
  int i = blockIdx.x * 256 + threadIdx.x;       // 2*64*4096
  int n = i & 4095, c = (i >> 12) & 63, b = i >> 18;
  fh[i] = (_Float16)g[((size_t)b * 65 + c) * 4096 + n];
}

// ---------------------------------------------------------------- GEMM 1 (WMMA)
// batt[b][n][m] = (sum_c fT[n][c]*fT[m][c]) * mask[n]*mask[m]
__global__ void k_batt(const _Float16* __restrict__ fT, const float* __restrict__ mask,
                       float* __restrict__ batt) {
  int lane = threadIdx.x & 31;
  int gw = blockIdx.x * 8 + (threadIdx.x >> 5);
  int b = gw >> 16;                // 65536 tiles/batch
  int rem = gw & 65535;
  int n0 = (rem >> 8) * 16, m0 = (rem & 255) * 16;
  const _Float16* base = fT + (size_t)b * 4096 * 64;
  v8f acc = {};
  acc = wmma_f16(load_A_frag(base, 64, n0, 0),  load_B_frag(base, 64, m0, 0),  acc);
  acc = wmma_f16(load_A_frag(base, 64, n0, 32), load_B_frag(base, 64, m0, 32), acc);
  int j = lane & 15, hi = lane >> 4;
  float mj = mask[b * 4096 + m0 + j];
  float* ob = batt + ((size_t)b * 4096 + n0) * 4096 + m0 + j;
#pragma unroll
  for (int v = 0; v < 8; ++v) {
    int M = v + 8 * hi;
    ob[(size_t)M * 4096] = acc[v] * mask[b * 4096 + n0 + M] * mj;
  }
}

// ---------------------------------------------------------------- softmax dim-1
__global__ void k_sm_part(const float* __restrict__ batt, float* __restrict__ part) {
  int m = blockIdx.x * 256 + threadIdx.x;       // column (coalesced)
  int b = blockIdx.y, seg = blockIdx.z;         // 16 n-segments of 256
  const float* p = batt + ((size_t)b * 4096 + seg * 256) * 4096 + m;
  float mx = -3.4e38f, s = 0.f;
  for (int n = 0; n < 256; ++n) {
    float v = p[(size_t)n * 4096];
    float nm = fmaxf(mx, v);
    s = s * __expf(mx - nm) + __expf(v - nm);
    mx = nm;
  }
  size_t o = ((size_t)b * 16 + seg) * 4096 + m;
  part[o * 2] = mx; part[o * 2 + 1] = s;
}
__global__ void k_sm_comb(const float* __restrict__ part, float* __restrict__ comb) {
  int i = blockIdx.x * 256 + threadIdx.x;       // b*4096+m, 8192 total
  int b = i >> 12, m = i & 4095;
  float mx = -3.4e38f;
  for (int s = 0; s < 16; ++s)
    mx = fmaxf(mx, part[(((size_t)b * 16 + s) * 4096 + m) * 2]);
  float sum = 0.f;
  for (int s = 0; s < 16; ++s) {
    size_t o = (((size_t)b * 16 + s) * 4096 + m) * 2;
    sum += part[o + 1] * __expf(part[o] - mx);
  }
  comb[(size_t)i * 2] = mx; comb[(size_t)i * 2 + 1] = 1.f / sum;
}
__global__ void k_sm_norm(float* __restrict__ batt, const float* __restrict__ comb) {
  int m = blockIdx.x * 256 + threadIdx.x;
  int b = blockIdx.y, seg = blockIdx.z;
  float mx = comb[((size_t)b * 4096 + m) * 2];
  float is = comb[((size_t)b * 4096 + m) * 2 + 1];
  float* p = batt + ((size_t)b * 4096 + seg * 256) * 4096 + m;
  for (int n = 0; n < 256; ++n)
    p[(size_t)n * 4096] = __expf(p[(size_t)n * 4096] - mx) * is;
}

// ---------------------------------------------------------------- avgpool3 -> f16
__global__ void k_pool3(const float* __restrict__ batt, _Float16* __restrict__ P) {
  size_t i = (size_t)blockIdx.x * 256 + threadIdx.x;   // 2*4096*4096
  int mm = (int)(i & 4095), n = (int)((i >> 12) & 4095), b = (int)(i >> 24);
  int y = mm >> 6, x = mm & 63;
  const float* p = batt + ((size_t)b * 4096 + n) * 4096;
  float s = 0.f;
#pragma unroll
  for (int dy = -1; dy <= 1; ++dy) {
    int yy = y + dy; if ((unsigned)yy >= 64u) continue;
#pragma unroll
    for (int dx = -1; dx <= 1; ++dx) {
      int xx = x + dx; if ((unsigned)xx >= 64u) continue;
      s += p[yy * 64 + xx];
    }
  }
  P[i] = (_Float16)(s * (1.f / 9.f));
}

// ---------------------------------------------------------------- GEMM 2 (WMMA)
// out_s[b][c][m] = sum_n A_s[b][c][n] * P[b][m][n], 4 sources sharing each B tile.
__global__ void k_att_apply(const _Float16* __restrict__ fh, const _Float16* __restrict__ x3h,
                            const _Float16* __restrict__ x21h, const _Float16* __restrict__ x11h,
                            const _Float16* __restrict__ P,
                            float* __restrict__ xo, float* __restrict__ x3a,
                            float* __restrict__ x2a, float* __restrict__ x1a) {
  int lane = threadIdx.x & 31;
  int gw = blockIdx.x * 8 + (threadIdx.x >> 5);   // B * 4 ctiles * 256 mtiles
  int b = gw >> 10, rem = gw & 1023;
  int c0 = (rem >> 8) * 16, m0 = (rem & 255) * 16;
  const _Float16* Pb = P + (size_t)b * 4096 * 4096;
  const _Float16* A0 = fh   + (size_t)b * 64 * 4096;
  const _Float16* A1 = x3h  + (size_t)b * 64 * 4096;
  const _Float16* A2 = x21h + (size_t)b * 64 * 4096;
  const _Float16* A3 = x11h + (size_t)b * 64 * 4096;
  v8f acc0 = {}, acc1 = {}, acc2 = {}, acc3 = {};
  for (int k0 = 0; k0 < 4096; k0 += 32) {
    v16h bf = load_B_frag(Pb, 4096, m0, k0);
    // prefetch next B chunk for this lane (global_prefetch_b8)
    __builtin_prefetch(Pb + (size_t)(m0 + (lane & 15)) * 4096 + k0 + 64, 0, 0);
    acc0 = wmma_f16(load_A_frag(A0, 4096, c0, k0), bf, acc0);
    acc1 = wmma_f16(load_A_frag(A1, 4096, c0, k0), bf, acc1);
    acc2 = wmma_f16(load_A_frag(A2, 4096, c0, k0), bf, acc2);
    acc3 = wmma_f16(load_A_frag(A3, 4096, c0, k0), bf, acc3);
  }
  int j = lane & 15, hi = lane >> 4;
  float* o0 = xo  + ((size_t)b * 64 + c0) * 4096 + m0 + j;
  float* o1 = x3a + ((size_t)b * 64 + c0) * 4096 + m0 + j;
  float* o2 = x2a + ((size_t)b * 64 + c0) * 4096 + m0 + j;
  float* o3 = x1a + ((size_t)b * 64 + c0) * 4096 + m0 + j;
#pragma unroll
  for (int v = 0; v < 8; ++v) {
    size_t r = (size_t)(v + 8 * hi) * 4096;
    o0[r] = acc0[v]; o1[r] = acc1[v]; o2[r] = acc2[v]; o3[r] = acc3[v];
  }
}

// ---------------------------------------------------------------- bilinear up2
__global__ void k_up2(const float* __restrict__ in, float* __restrict__ out,
                      int CT, int h, int w) {   // out [CT, 2h, 2w], half-pixel
  size_t i = (size_t)blockIdx.x * 256 + threadIdx.x;
  size_t total = (size_t)CT * 4 * h * w;
  if (i >= total) return;
  int ow = 2 * w, oh = 2 * h;
  int x = (int)(i % ow);
  int y = (int)((i / ow) % oh);
  int c = (int)(i / ((size_t)ow * oh));
  float sy = y * 0.5f - 0.25f, sx = x * 0.5f - 0.25f;
  int y0f = (int)floorf(sy), x0f = (int)floorf(sx);
  float wy = sy - (float)y0f, wx = sx - (float)x0f;
  int y0 = min(max(y0f, 0), h - 1), y1 = min(max(y0f + 1, 0), h - 1);
  int x0 = min(max(x0f, 0), w - 1), x1 = min(max(x0f + 1, 0), w - 1);
  const float* p = in + (size_t)c * h * w;
  float v = (1.f - wy) * ((1.f - wx) * p[(size_t)y0 * w + x0] + wx * p[(size_t)y0 * w + x1]) +
            wy         * ((1.f - wx) * p[(size_t)y1 * w + x0] + wx * p[(size_t)y1 * w + x1]);
  out[i] = v;
}

// out = a*mask + (1-mask)*alt  over [B,64,4096]; `a` has channel stride aCS rows.
__global__ void k_combine(const float* __restrict__ a, int aCS,
                          const float* __restrict__ alt, const float* __restrict__ mask,
                          float* __restrict__ out) {
  int i = blockIdx.x * 256 + threadIdx.x;   // 2*64*4096
  int p = i & 4095, c = (i >> 12) & 63, b = i >> 18;
  float m = mask[b * 4096 + p];
  float av = a[((size_t)b * aCS + c) * 4096 + p];
  out[i] = av * m + (1.f - m) * alt[i];
}

// out = xin*m + (1-m)*relu(conv1x1(xup))  (used for o1, o2)
__global__ void k_out_conv(const float* __restrict__ xin, const float* __restrict__ xup,
                           const float* __restrict__ mfull, const float* __restrict__ w,
                           const float* __restrict__ bias, float* __restrict__ out,
                           int Cin, int Cout, int HW) {
  int p = blockIdx.x * 256 + threadIdx.x;
  if (p >= HW) return;
  int bc = blockIdx.y, b = bc / Cout, co = bc % Cout;
  const float* ip = xup + (size_t)b * Cin * HW + p;
  const float* wp = w + (size_t)co * Cin;
  float acc = bias[co];
  for (int ci = 0; ci < Cin; ++ci) acc = fmaf(wp[ci], ip[(size_t)ci * HW], acc);
  acc = fmaxf(acc, 0.f);
  float m = mfull[(size_t)b * HW + p];
  out[(size_t)bc * HW + p] = xin[(size_t)bc * HW + p] * m + (1.f - m) * acc;
}

// ================================================================ launch
extern "C" void kernel_launch(void* const* d_in, const int* in_sizes, int n_in,
                              void* d_out, int out_size, void* d_ws, size_t ws_size,
                              hipStream_t stream) {
  (void)in_sizes; (void)n_in; (void)out_size; (void)ws_size;
  const float* x1  = (const float*)d_in[0];
  const float* x2  = (const float*)d_in[1];
  const float* x3  = (const float*)d_in[2];
  const float* xm  = (const float*)d_in[3];
  const float* wg1 = (const float*)d_in[4];  const float* bg1 = (const float*)d_in[5];
  const float* wg2 = (const float*)d_in[6];  const float* bg2 = (const float*)d_in[7];
  const float* wg3 = (const float*)d_in[8];  const float* bg3 = (const float*)d_in[9];
  const float* wc  = (const float*)d_in[10]; const float* bcb = (const float*)d_in[11];
  const float* wc2 = (const float*)d_in[12]; const float* bc2 = (const float*)d_in[13];
  const float* wc3 = (const float*)d_in[14]; const float* bc3 = (const float*)d_in[15];
  const float* wc4 = (const float*)d_in[16]; const float* bc4 = (const float*)d_in[17];
  const float* wc5 = (const float*)d_in[18]; const float* bc5 = (const float*)d_in[19];

  // outputs (concat in return order, all f32)
  float* o1 = (float*)d_out;                 // [2,32,256,256]
  float* o2 = o1 + 4194304;                  // [2,64,128,128]
  float* o3 = o2 + 2097152;                  // [2,64,64,64]
  float* oo = o3 + 524288;                   // [2,64,64,64]
  float* m4 = oo + 524288;                   // [2,1,256,256]

  // workspace bump allocator
  char* wsb = (char*)d_ws;
  size_t off = 0;
  auto alloc = [&](size_t bytes) -> void* {
    void* p = wsb + off;
    off = (off + bytes + 255) & ~(size_t)255;
    return p;
  };
  float*    batt   = (float*)   alloc((size_t)2 * 4096 * 4096 * 4);  // 134 MB
  _Float16* P      = (_Float16*)alloc((size_t)2 * 4096 * 4096 * 2);  // 67 MB
  float*    x1p    = (float*)   alloc((size_t)2 * 32 * 4096 * 4);
  float*    x2p    = (float*)   alloc((size_t)2 * 64 * 4096 * 4);
  float*    x1_1   = (float*)   alloc((size_t)2 * 64 * 4096 * 4);
  float*    x2_1   = (float*)   alloc((size_t)2 * 64 * 4096 * 4);
  float*    g0     = (float*)   alloc((size_t)2 * 65 * 4096 * 4);
  float*    g1     = (float*)   alloc((size_t)2 * 65 * 4096 * 4);
  float*    yt     = (float*)   alloc((size_t)2 * 65 * 4096 * 4);
  _Float16* fT     = (_Float16*)alloc((size_t)2 * 4096 * 64 * 2);
  _Float16* fh     = (_Float16*)alloc((size_t)2 * 64 * 4096 * 2);
  _Float16* x3h    = (_Float16*)alloc((size_t)2 * 64 * 4096 * 2);
  _Float16* x21h   = (_Float16*)alloc((size_t)2 * 64 * 4096 * 2);
  _Float16* x11h   = (_Float16*)alloc((size_t)2 * 64 * 4096 * 2);
  float*    smp    = (float*)   alloc((size_t)2 * 16 * 4096 * 2 * 4);
  float*    smc    = (float*)   alloc((size_t)2 * 4096 * 2 * 4);
  float*    xo     = (float*)   alloc((size_t)2 * 64 * 4096 * 4);
  float*    x3a    = (float*)   alloc((size_t)2 * 64 * 4096 * 4);
  float*    x2a64  = (float*)   alloc((size_t)2 * 64 * 4096 * 4);
  float*    x1a64  = (float*)   alloc((size_t)2 * 64 * 4096 * 4);
  float*    x2aup  = (float*)   alloc((size_t)2 * 64 * 16384 * 4);
  float*    x1aup1 = (float*)   alloc((size_t)2 * 64 * 16384 * 4);
  float*    x1aup2 = (float*)   alloc((size_t)2 * 64 * 65536 * 4);
  float*    m2     = (float*)   alloc((size_t)2 * 16384 * 4);

  const dim3 B256(256);

  // --- front-end: pools + 1x1 convs -> x1_1, x2_1, g0
  k_maxpool4<<<dim3((2 * 32 * 4096) / 256), B256, 0, stream>>>(x1, x1p, 64, 64, 64);
  k_conv1x1 <<<dim3(16, 128), B256, 0, stream>>>(x1p, wc4, bc4, x1_1, 32, 64, 4096);
  k_maxpool2<<<dim3((2 * 64 * 4096) / 256), B256, 0, stream>>>(x2, x2p, 128, 64, 64);
  k_conv1x1 <<<dim3(16, 128), B256, 0, stream>>>(x2p, wc5, bc5, x2_1, 64, 64, 4096);
  k_f1      <<<dim3(16, 130), B256, 0, stream>>>(x1_1, x2_1, x3, wc, bcb, xm, g0);

  // --- gate conv chain (3x3 -> 5x5 -> 7x7)
  k_conv_direct<<<dim3(16, 130), B256, 0, stream>>>(g0, wg3, bg3, yt, 3, 1);
  k_gate       <<<dim3(16, 130), B256, 0, stream>>>(yt, g1);
  k_conv_direct<<<dim3(16, 130), B256, 0, stream>>>(g1, wg2, bg2, yt, 5, 2);
  k_gate       <<<dim3(16, 130), B256, 0, stream>>>(yt, g0);        // g0 := xa2
  k_conv_direct<<<dim3(16, 130), B256, 0, stream>>>(g0, wg1, bg1, yt, 7, 3);
  k_gate       <<<dim3(16, 130), B256, 0, stream>>>(yt, g1);        // g1 := f (+mask ch)

  // --- f16 staging
  k_fT     <<<dim3(2048), B256, 0, stream>>>(g1, fT);
  k_fh     <<<dim3(2048), B256, 0, stream>>>(g1, fh);
  k_to_half<<<dim3(2048), B256, 0, stream>>>(x3,   x3h,  2 * 64 * 4096);
  k_to_half<<<dim3(2048), B256, 0, stream>>>(x2_1, x21h, 2 * 64 * 4096);
  k_to_half<<<dim3(2048), B256, 0, stream>>>(x1_1, x11h, 2 * 64 * 4096);

  // --- attention matrix: WMMA GEMM, softmax(dim=1), 3x3 avgpool -> f16
  k_batt   <<<dim3(16384), B256, 0, stream>>>(fT, xm, batt);
  k_sm_part<<<dim3(16, 2, 16), B256, 0, stream>>>(batt, smp);
  k_sm_comb<<<dim3(32), B256, 0, stream>>>(smp, smc);
  k_sm_norm<<<dim3(16, 2, 16), B256, 0, stream>>>(batt, smc);
  k_pool3  <<<dim3(131072), B256, 0, stream>>>(batt, P);

  // --- apply attention to 4 sources in one fused WMMA kernel
  k_att_apply<<<dim3(256), B256, 0, stream>>>(fh, x3h, x21h, x11h, P,
                                              xo, x3a, x2a64, x1a64);

  // --- outputs at 64x64
  k_combine<<<dim3(2048), B256, 0, stream>>>(g1, 65, xo,  xm, oo);   // out
  k_combine<<<dim3(2048), B256, 0, stream>>>(x3, 64, x3a, xm, o3);   // o3

  // --- masks: m2 = up2(mask), m4 = up2(m2) (m4 is also an output)
  k_up2<<<dim3(128), B256, 0, stream>>>(xm, m2, 2, 64, 64);
  k_up2<<<dim3(512), B256, 0, stream>>>(m2, m4, 2, 128, 128);

  // --- o2 branch: up2 -> conv1x1+relu -> combine
  k_up2     <<<dim3(8192), B256, 0, stream>>>(x2a64, x2aup, 128, 64, 64);
  k_out_conv<<<dim3(64, 128), B256, 0, stream>>>(x2, x2aup, m2, wc2, bc2, o2,
                                                 64, 64, 16384);

  // --- o1 branch: up2 -> up2 -> conv1x1+relu -> combine
  k_up2     <<<dim3(8192),  B256, 0, stream>>>(x1a64,  x1aup1, 128, 64, 64);
  k_up2     <<<dim3(32768), B256, 0, stream>>>(x1aup1, x1aup2, 128, 128, 128);
  k_out_conv<<<dim3(256, 64), B256, 0, stream>>>(x1, x1aup2, m4, wc3, bc3, o1,
                                                 64, 32, 65536);
}
